// OctreeConvBnRelu_24988119728282
// MI455X (gfx1250) — matrix operand order repaired and checked by reading, not compiled, not run on hardware
//
#include <hip/hip_runtime.h>
#include <hip/hip_bf16.h>

// CDNA5 WMMA vector types
typedef __attribute__((ext_vector_type(16))) __bf16 v16bf;
typedef __attribute__((ext_vector_type(8)))  float  v8f;

#define KSTEN 27   // stencil taps (divisible by 3: 9 pipeline iterations)
#define CIN   32   // input channels  == WMMA K for bf16 16x16x32
#define COUT  64   // output channels == 4 WMMA N-tiles
#define BN_EPS 1e-5f

// Fragment-layout LDS weight cache: [k][t][lane][e], e=0..15 bf16 per lane.
// B layout used here (contiguous k, same as A): lane holds column N=lane%16; element e holds
// K = e + (lane>=16 ? 16 : 0).
#define WLDS_ELEMS (KSTEN * 4 * 32 * 16)   // 55296 bf16 = 108 KB

__global__ __launch_bounds__(256) void octconv_wmma_kernel(
    const float* __restrict__ data,    // [N, 32]
    const int*   __restrict__ neigh,   // [N, 27]
    const float* __restrict__ weight,  // [27, 32, 64]
    float*       __restrict__ out,     // [N, 64] (raw conv result)
    float*       __restrict__ gacc,    // [128]: sum[64], sumsq[64] (pre-zeroed)
    int N)
{
    __shared__ __bf16 lds_w[WLDS_ELEMS];
    __shared__ float  bsum[COUT];
    __shared__ float  bsq[COUT];

    const int tid = threadIdx.x;
    if (tid < COUT) { bsum[tid] = 0.0f; bsq[tid] = 0.0f; }

    // ---- Stage weights fp32 -> bf16 into WMMA B-fragment layout ----
    // Coalesced float4 loads, 9 in flight per thread, then scatter to the
    // swizzled LDS layout.  6 passes x 9 x 256 threads x 4 floats = 55296.
    {
        const float4* wq = (const float4*)weight;
        #pragma unroll 1
        for (int p = 0; p < 6; ++p) {
            float4 v[9];
            #pragma unroll
            for (int u = 0; u < 9; ++u)
                v[u] = wq[(p * 9 + u) * 256 + tid];   // 9 b128 loads in flight
            #pragma unroll
            for (int u = 0; u < 9; ++u) {
                const int f   = (p * 9 + u) * 256 + tid; // float4 index in weight
                const int col = (f & 15) << 2;           // output channel base
                const int K   = (f >> 4) & 31;           // input channel
                const int k   = f >> 9;                  // stencil tap
                const int t   = col >> 4;                // WMMA N-tile
                const int e   = K & 15;                  // fragment element
                const int lh  = K & 16;                  // lane-half offset
                const int fb  = ((k * 4 + t) * 32 + lh) * 16 + e;
                const float s4[4] = {v[u].x, v[u].y, v[u].z, v[u].w};
                #pragma unroll
                for (int j = 0; j < 4; ++j)
                    lds_w[fb + ((col + j) & 15) * 16] = (__bf16)s4[j];
            }
        }
    }
    __syncthreads();

    const int lane          = tid & 31;
    const int wavesPerBlock = 256 >> 5;
    const int waveId        = blockIdx.x * wavesPerBlock + (tid >> 5);
    const int nWaves        = gridDim.x * wavesPerBlock;
    const int numTiles      = (N + 15) >> 4;

    const int m    = lane & 15;                // A row this lane supplies
    const int c0   = (lane & 16) ? 16 : 0;     // channel chunk base (A layout)
    const int mrow = (lane & 16) ? 8 : 0;      // D-fragment row base

    float psum[4] = {0.f, 0.f, 0.f, 0.f};
    float psq[4]  = {0.f, 0.f, 0.f, 0.f};

    for (int tile = waveId; tile < numTiles; tile += nWaves) {
        const int base = tile << 4;
        int r = base + m;
        const bool rowOK = (r < N);
        if (!rowOK) r = N - 1;
        const int* nrow = neigh + (size_t)r * KSTEN;

        v8f acc[4];
        #pragma unroll
        for (int t = 0; t < 4; ++t)
            acc[t] = (v8f){0.f,0.f,0.f,0.f,0.f,0.f,0.f,0.f};

        auto gatherIssue = [&](float4* b, int idx) {
            if (idx < 0) idx = 0;
            const float* dp = data + (size_t)idx * CIN + c0;
            b[0] = *(const float4*)(dp);
            b[1] = *(const float4*)(dp + 4);
            b[2] = *(const float4*)(dp + 8);
            b[3] = *(const float4*)(dp + 12);
        };
        auto consume = [&](const float4* bf, bool valid, int k) {
            float4 f0 = bf[0], f1 = bf[1], f2 = bf[2], f3 = bf[3];
            if (!valid) {  // closes before WMMA; EXEC full at WMMA issue
                f0 = make_float4(0.f,0.f,0.f,0.f); f1 = f0; f2 = f0; f3 = f0;
            }
            // A fragment (16x32 bf16): e=0..15 -> ch c0+e
            v16bf a;
            a[0]=(__bf16)f0.x;  a[1]=(__bf16)f0.y;  a[2]=(__bf16)f0.z;  a[3]=(__bf16)f0.w;
            a[4]=(__bf16)f1.x;  a[5]=(__bf16)f1.y;  a[6]=(__bf16)f1.z;  a[7]=(__bf16)f1.w;
            a[8]=(__bf16)f2.x;  a[9]=(__bf16)f2.y;  a[10]=(__bf16)f2.z; a[11]=(__bf16)f2.w;
            a[12]=(__bf16)f3.x; a[13]=(__bf16)f3.y; a[14]=(__bf16)f3.z; a[15]=(__bf16)f3.w;

            const int wbase = ((k << 2) * 32 + lane) << 4;  // 32B-aligned
            #pragma unroll
            for (int t = 0; t < 4; ++t) {
                v16bf b = *(const v16bf*)&lds_w[wbase + (t << 9)];
                // (neg_a, A, neg_b, B, c_mod, C, reuse_a, reuse_b)
                acc[t] = __builtin_amdgcn_wmma_f32_16x16x32_bf16(
                    false, a, false, b, (short)0, acc[t], false, false);
            }
        };

        // Depth-2 software pipeline over the 27 taps, written as 9 iterations
        // of 3 explicit stages so the triple buffer uses static register names
        // (bounded VGPR pressure, no dynamic register-array indexing).
        float4 b0[4], b1[4], b2[4];
        int ia = nrow[0], ib = nrow[1], ic = nrow[2], id = nrow[3];
        gatherIssue(b0, ia);
        gatherIssue(b1, ib);

        #pragma unroll 1
        for (int kb = 0; kb < KSTEN; kb += 3) {
            const int ie = (kb + 4 < KSTEN) ? nrow[kb + 4] : -1;
            gatherIssue(b2, ic);                 // tap kb+2 in flight
            consume(b0, rowOK && ia >= 0, kb);

            const int if_ = (kb + 5 < KSTEN) ? nrow[kb + 5] : -1;
            gatherIssue(b0, id);                 // tap kb+3 in flight
            consume(b1, rowOK && ib >= 0, kb + 1);

            const int ig = (kb + 6 < KSTEN) ? nrow[kb + 6] : -1;
            gatherIssue(b1, ie);                 // tap kb+4 in flight
            consume(b2, rowOK && ic >= 0, kb + 2);

            ia = id; ib = ie; ic = if_; id = ig; // rotate index pipeline
        }

        // D layout: lane gives column (t*16 + lane%16); VGPR j -> row mrow+j.
        #pragma unroll
        for (int t = 0; t < 4; ++t) {
            const int col = (t << 4) + (lane & 15);
            #pragma unroll
            for (int j = 0; j < 8; ++j) {
                const float v = acc[t][j];
                const int row = base + mrow + j;
                if (row < N) out[(size_t)row * COUT + col] = v;
                psum[t] += v;            // OOB rows contribute exact 0
                psq[t]  += v * v;
            }
        }
    }

    // ---- BN moment reduction: lane-private -> LDS -> global atomics ----
    #pragma unroll
    for (int t = 0; t < 4; ++t) {
        const int c = (t << 4) + (lane & 15);
        atomicAdd(&bsum[c], psum[t]);
        atomicAdd(&bsq[c],  psq[t]);
    }
    __syncthreads();
    if (tid < COUT) {
        unsafeAtomicAdd(&gacc[tid],        bsum[tid]);
        unsafeAtomicAdd(&gacc[COUT + tid], bsq[tid]);
    }
}

__global__ __launch_bounds__(256) void bnrelu_kernel(
    float* __restrict__ out,          // [N, 64] in place
    const float* __restrict__ gacc,   // sum[64], sumsq[64]
    const float* __restrict__ gamma,
    const float* __restrict__ beta,
    int total4, float invN)
{
    const int i = blockIdx.x * blockDim.x + threadIdx.x;
    if (i >= total4) return;
    float4 v = ((const float4*)out)[i];
    float r[4] = {v.x, v.y, v.z, v.w};
    const int cbase = (i << 2) & (COUT - 1);   // 4 consecutive channels, same row
    #pragma unroll
    for (int u = 0; u < 4; ++u) {
        const int c = cbase + u;
        const float mean = gacc[c] * invN;
        const float var  = gacc[COUT + c] * invN - mean * mean;
        const float s    = gamma[c] * rsqrtf(var + BN_EPS);
        float val = (r[u] - mean) * s + beta[c];
        r[u] = val > 0.0f ? val : 0.0f;
    }
    ((float4*)out)[i] = make_float4(r[0], r[1], r[2], r[3]);
}

extern "C" void kernel_launch(void* const* d_in, const int* in_sizes, int n_in,
                              void* d_out, int out_size, void* d_ws, size_t ws_size,
                              hipStream_t stream) {
    const float* data   = (const float*)d_in[0];
    const int*   neigh  = (const int*)d_in[1];   // JAX x64-off: int64 -> int32
    const float* weight = (const float*)d_in[2];
    const float* gamma  = (const float*)d_in[3];
    const float* beta   = (const float*)d_in[4];
    float*       out    = (float*)d_out;
    float*       gacc   = (float*)d_ws;          // 128 floats

    const int N = in_sizes[0] / CIN;

    hipMemsetAsync(gacc, 0, 2 * COUT * sizeof(float), stream);

    const int numTiles = (N + 15) / 16;
    int blocks = (numTiles + 7) / 8;             // 8 waves/block, 1 tile/wave min
    if (blocks > 512) blocks = 512;              // amortize LDS weight staging
    if (blocks < 1) blocks = 1;
    octconv_wmma_kernel<<<blocks, 256, 0, stream>>>(data, neigh, weight, out, gacc, N);

    const int total4 = (N * COUT) / 4;
    bnrelu_kernel<<<(total4 + 255) / 256, 256, 0, stream>>>(
        out, gacc, gamma, beta, total4, 1.0f / (float)N);
}
